// Decoder_6957847019934
// MI455X (gfx1250) — matrix operand order repaired and checked by reading, not compiled, not run on hardware
//
#include <hip/hip_runtime.h>
#include <hip/hip_bf16.h>

// ---------------------------------------------------------------------------
// CDNA5 (gfx1250) autoregressive GRU decoder.
// bf16 weights/activations, fp32 accumulate via v_wmma_f32_16x16x32_bf16.
// Weights (~33MB bf16) are L2-resident (192MB L2).  Each wave32 computes a
// 64x16 slab of h_new (4 m-tiles register-blocked against one n-tile): each
// weight fragment is loaded once per 4 tiles -> 12 independent WMMA chains per
// K-chunk.  All global addresses are uniform-base + 32-bit lane offsets so the
// backend can use SADDR-mode loads with immediate K offsets (minimal VALU).
// ---------------------------------------------------------------------------

typedef __bf16 bf16;
typedef __attribute__((ext_vector_type(16))) __bf16 v16bf;
typedef __attribute__((ext_vector_type(8)))  __bf16 v8bf;
typedef __attribute__((ext_vector_type(8)))  float  v8f;

#define GRU_H   1024
#define GRU_B   512
#define GRU_T   128
#define GRU_V   64
#define GRU_LAT 256
#define KX0PAD  352   // LAT + V + CON = 321 padded to multiple of 32
#define MTILES  4     // m-tiles (16 rows each) per wave

union AFrag { v16bf v; v8bf h[2]; };

__device__ __forceinline__ float sigf(float x) {
    return 1.0f / (1.0f + __expf(-x));
}
__device__ __forceinline__ float tanh_fast(float x) {
    float e = __expf(2.0f * x);
    return (e - 1.0f) / (e + 1.0f);
}

// 32-byte contiguous per-lane load: uniform base + 32-bit byte offset.
__device__ __forceinline__ v16bf load16bf(const bf16* __restrict__ base,
                                          unsigned byteoff) {
    return *(const v16bf*)((const char*)base + byteoff);
}
__device__ __forceinline__ v8bf load8bf(const bf16* __restrict__ base,
                                        unsigned byteoff) {
    return *(const v8bf*)((const char*)base + byteoff);
}

// One wave -> 64x16 slab of h_new (4 m-tiles x 1 n-tile).
// grid: 64 blocks x 256 threads (8 waves/block, 512 waves =
// 8 m-groups * 64 n-tiles).
//
// Fragment layouts (wave32, v_wmma_f32_16x16x32_bf16):
//   A 16x32: M = lane&15; elems 0..7 = K[ksel*16..], elems 8..15 = K[ksel*16+8..]
//            -> two contiguous 16B loads per lane.
//   B 32x16: N = lane&15 (B = W^T, so column n == weight row n);
//            elems 0..15 = K[ksel*16..] -> one contiguous 32B load per lane.
//   C/D:     element i -> row = i + (lane>=16)*8, col = lane&15.
template <int KX>
__global__ __launch_bounds__(256) void gru_layer_kernel(
    const bf16* __restrict__ x,
    const bf16* __restrict__ wih, const bf16* __restrict__ whh,
    const float* __restrict__ bih, const float* __restrict__ bhh,
    const bf16* __restrict__ hin, bf16* __restrict__ hout)
{
    const int lane = threadIdx.x & 31;
    const int wid  = blockIdx.x * 8 + (threadIdx.x >> 5);   // 0..511
    const int m_base = (wid & 7) * (16 * MTILES);           // over B = 512
    const int n_base = (wid >> 3) << 4;                     // over H = 1024
    const int nl   = lane & 15;
    const int ksel = lane >> 4;

    // Per-lane 32-bit byte offsets (all buffers < 4GB).
    unsigned aoffX[MTILES], aoffH[MTILES];
#pragma unroll
    for (int mt = 0; mt < MTILES; ++mt) {
        aoffX[mt] = (unsigned)(((m_base + mt * 16 + nl) * KX    + ksel * 16) * 2);
        aoffH[mt] = (unsigned)(((m_base + mt * 16 + nl) * GRU_H + ksel * 16) * 2);
    }
    unsigned boffX[3], boffH[3];
#pragma unroll
    for (int g = 0; g < 3; ++g) {
        boffX[g] = (unsigned)(((g * GRU_H + n_base + nl) * KX    + ksel * 16) * 2);
        boffH[g] = (unsigned)(((g * GRU_H + n_base + nl) * GRU_H + ksel * 16) * 2);
    }

    v8f accR[MTILES]  = {};
    v8f accZ[MTILES]  = {};
    v8f accXN[MTILES] = {};
    v8f accHN[MTILES] = {};

    // ---- gx contributions: r, z, xn ----
#pragma unroll 4
    for (int kb = 0; kb < KX; kb += 32) {
        const unsigned kbb = (unsigned)(kb * 2);
        v16bf a[MTILES];
#pragma unroll
        for (int mt = 0; mt < MTILES; ++mt) {
            AFrag af;
            af.h[0] = load8bf(x, aoffX[mt] + kbb);
            af.h[1] = load8bf(x, aoffX[mt] + kbb + 16);
            a[mt] = af.v;
        }
        const v16bf br = load16bf(wih, boffX[0] + kbb);
        const v16bf bz = load16bf(wih, boffX[1] + kbb);
        const v16bf bn = load16bf(wih, boffX[2] + kbb);
#pragma unroll
        for (int mt = 0; mt < MTILES; ++mt) {
            accR[mt]  = __builtin_amdgcn_wmma_f32_16x16x32_bf16(false, a[mt], false, br,
                            (short)0, accR[mt],  false, false);
            accZ[mt]  = __builtin_amdgcn_wmma_f32_16x16x32_bf16(false, a[mt], false, bz,
                            (short)0, accZ[mt],  false, false);
            accXN[mt] = __builtin_amdgcn_wmma_f32_16x16x32_bf16(false, a[mt], false, bn,
                            (short)0, accXN[mt], false, false);
        }
    }
    // ---- gh contributions: r, z, hn ----
#pragma unroll 4
    for (int kb = 0; kb < GRU_H; kb += 32) {
        const unsigned kbb = (unsigned)(kb * 2);
        v16bf a[MTILES];
#pragma unroll
        for (int mt = 0; mt < MTILES; ++mt) {
            AFrag af;
            af.h[0] = load8bf(hin, aoffH[mt] + kbb);
            af.h[1] = load8bf(hin, aoffH[mt] + kbb + 16);
            a[mt] = af.v;
        }
        const v16bf br = load16bf(whh, boffH[0] + kbb);
        const v16bf bz = load16bf(whh, boffH[1] + kbb);
        const v16bf bn = load16bf(whh, boffH[2] + kbb);
#pragma unroll
        for (int mt = 0; mt < MTILES; ++mt) {
            accR[mt]  = __builtin_amdgcn_wmma_f32_16x16x32_bf16(false, a[mt], false, br,
                            (short)0, accR[mt],  false, false);
            accZ[mt]  = __builtin_amdgcn_wmma_f32_16x16x32_bf16(false, a[mt], false, bz,
                            (short)0, accZ[mt],  false, false);
            accHN[mt] = __builtin_amdgcn_wmma_f32_16x16x32_bf16(false, a[mt], false, bn,
                            (short)0, accHN[mt], false, false);
        }
    }

    // ---- epilogue: gates + h update ----
    const int ncol  = n_base + nl;
    const float biasR = bih[ncol] + bhh[ncol];
    const float biasZ = bih[GRU_H + ncol] + bhh[GRU_H + ncol];
    const float bxn   = bih[2 * GRU_H + ncol];
    const float bhn   = bhh[2 * GRU_H + ncol];
#pragma unroll
    for (int mt = 0; mt < MTILES; ++mt) {
#pragma unroll
        for (int i = 0; i < 8; ++i) {
            const int row = m_base + mt * 16 + i + (ksel << 3);
            const unsigned hoff = (unsigned)((row * GRU_H + ncol) * 2);
            const float hold = (float)*(const bf16*)((const char*)hin + hoff);
            const float r = sigf(accR[mt][i] + biasR);
            const float z = sigf(accZ[mt][i] + biasZ);
            const float n = tanh_fast(accXN[mt][i] + bxn + r * (accHN[mt][i] + bhn));
            *(bf16*)((char*)hout + hoff) = (bf16)((1.0f - z) * n + z * hold);
        }
    }
}

// FC(64 x 1024) + argmax + one-hot feedback.  One block (64 threads) per row.
__global__ __launch_bounds__(64) void fc_argmax_kernel(
    const bf16* __restrict__ h2, const float* __restrict__ fcw,
    const float* __restrict__ fcb, float* __restrict__ out,
    bf16* __restrict__ x0, int t)
{
    __shared__ float logits[GRU_V];
    __shared__ int sel;
    const int b = blockIdx.x;
    const int j = threadIdx.x;
    const bf16*  hr = h2  + (size_t)b * GRU_H;
    const float* wr = fcw + (size_t)j * GRU_H;
    float acc = fcb[j];
#pragma unroll 4
    for (int k = 0; k < GRU_H; k += 4) {
        acc += (float)hr[k]     * wr[k]
             + (float)hr[k + 1] * wr[k + 1]
             + (float)hr[k + 2] * wr[k + 2]
             + (float)hr[k + 3] * wr[k + 3];
    }
    logits[j] = acc;
    __syncthreads();
    if (j == 0) {                 // first-max tie-break, matches jnp.argmax
        int s = 0; float best = logits[0];
        for (int i = 1; i < GRU_V; ++i)
            if (logits[i] > best) { best = logits[i]; s = i; }
        sel = s;
    }
    __syncthreads();
    const float oh = (j == sel) ? 1.0f : 0.0f;
    out[((size_t)b * GRU_T + t) * GRU_V + j] = oh;
    // feedback: x_next = [latent | onehot | 0]; zero the CON slot too
    x0[(size_t)b * KX0PAD + GRU_LAT + j] = (bf16)oh;
    if (j == 0) x0[(size_t)b * KX0PAD + GRU_LAT + GRU_V] = (bf16)0.0f;
}

// fp32 -> bf16 with optional K-padding (zeros past Ksrc).
__global__ void convert_pad_kernel(const float* __restrict__ src,
                                   bf16* __restrict__ dst,
                                   int rows, int Ksrc, int Kdst)
{
    size_t idx = (size_t)blockIdx.x * 256 + threadIdx.x;
    size_t total = (size_t)rows * Kdst;
    if (idx >= total) return;
    int r = (int)(idx / Kdst);
    int k = (int)(idx % Kdst);
    dst[idx] = (k < Ksrc) ? (bf16)src[(size_t)r * Ksrc + k] : (bf16)0.0f;
}

// x0 = [latent(256) | zeros(64) | enthalpy | zeros(pad)]
__global__ void init_x0_kernel(const float* __restrict__ latent,
                               const float* __restrict__ enth,
                               bf16* __restrict__ x0)
{
    size_t idx = (size_t)blockIdx.x * 256 + threadIdx.x;
    if (idx >= (size_t)GRU_B * KX0PAD) return;
    int b = (int)(idx / KX0PAD);
    int k = (int)(idx % KX0PAD);
    float v = 0.0f;
    if (k < GRU_LAT)                       v = latent[(size_t)b * GRU_LAT + k];
    else if (k == GRU_LAT + GRU_V)         v = enth[b];
    x0[idx] = (bf16)v;
}

__global__ void zero_u32_kernel(unsigned int* __restrict__ p, size_t n)
{
    size_t idx = (size_t)blockIdx.x * 256 + threadIdx.x;
    if (idx < n) p[idx] = 0u;
}

extern "C" void kernel_launch(void* const* d_in, const int* in_sizes, int n_in,
                              void* d_out, int out_size, void* d_ws, size_t ws_size,
                              hipStream_t stream)
{
    (void)in_sizes; (void)n_in; (void)out_size; (void)ws_size;
    // setup_inputs() order: latent_vec, enthalpy, inp,
    //   params{ layers[3]{w_ih,w_hh,b_ih,b_hh}, fc_w, fc_b }
    const float* latent = (const float*)d_in[0];
    const float* enth   = (const float*)d_in[1];
    // d_in[2] = inp (unused on the freerun path)
    const float* wih_f[3] = {(const float*)d_in[3], (const float*)d_in[7],  (const float*)d_in[11]};
    const float* whh_f[3] = {(const float*)d_in[4], (const float*)d_in[8],  (const float*)d_in[12]};
    const float* bih[3]   = {(const float*)d_in[5], (const float*)d_in[9],  (const float*)d_in[13]};
    const float* bhh[3]   = {(const float*)d_in[6], (const float*)d_in[10], (const float*)d_in[14]};
    const float* fcw = (const float*)d_in[15];
    const float* fcb = (const float*)d_in[16];
    float* out = (float*)d_out;

    // workspace bump allocator (256B aligned regions)
    char* ws = (char*)d_ws;
    size_t off = 0;
    auto wsalloc = [&](size_t bytes) -> void* {
        void* p = ws + off;
        off = (off + bytes + 255) & ~(size_t)255;
        return p;
    };
    bf16* wih_b[3];
    wih_b[0] = (bf16*)wsalloc((size_t)3 * GRU_H * KX0PAD * 2);
    wih_b[1] = (bf16*)wsalloc((size_t)3 * GRU_H * GRU_H * 2);
    wih_b[2] = (bf16*)wsalloc((size_t)3 * GRU_H * GRU_H * 2);
    bf16* whh_b[3];
    for (int l = 0; l < 3; ++l)
        whh_b[l] = (bf16*)wsalloc((size_t)3 * GRU_H * GRU_H * 2);
    bf16* x0 = (bf16*)wsalloc((size_t)GRU_B * KX0PAD * 2);
    bf16* hstate = (bf16*)wsalloc((size_t)3 * 2 * GRU_B * GRU_H * 2); // [layer][parity]
    bf16* hbuf[3][2];
    for (int l = 0; l < 3; ++l)
        for (int p = 0; p < 2; ++p)
            hbuf[l][p] = hstate + ((size_t)l * 2 + p) * GRU_B * GRU_H;

    // --- init: weight conversion, x0, zero hidden state ---
    auto conv = [&](const float* s, bf16* d, int rows, int Ks, int Kd) {
        size_t total = (size_t)rows * Kd;
        convert_pad_kernel<<<(unsigned)((total + 255) / 256), 256, 0, stream>>>(s, d, rows, Ks, Kd);
    };
    conv(wih_f[0], wih_b[0], 3 * GRU_H, GRU_LAT + GRU_V + 1, KX0PAD);
    conv(wih_f[1], wih_b[1], 3 * GRU_H, GRU_H, GRU_H);
    conv(wih_f[2], wih_b[2], 3 * GRU_H, GRU_H, GRU_H);
    for (int l = 0; l < 3; ++l)
        conv(whh_f[l], whh_b[l], 3 * GRU_H, GRU_H, GRU_H);

    {
        size_t n = (size_t)GRU_B * KX0PAD;
        init_x0_kernel<<<(unsigned)((n + 255) / 256), 256, 0, stream>>>(latent, enth, x0);
    }
    {
        size_t words = (size_t)3 * 2 * GRU_B * GRU_H * 2 / 4;
        zero_u32_kernel<<<(unsigned)((words + 255) / 256), 256, 0, stream>>>(
            (unsigned int*)hstate, words);
    }

    // --- 128 sequential decode steps ---
    for (int t = 0; t < GRU_T; ++t) {
        const int p = t & 1;
        gru_layer_kernel<KX0PAD><<<64, 256, 0, stream>>>(
            x0, wih_b[0], whh_b[0], bih[0], bhh[0],
            hbuf[0][p], hbuf[0][p ^ 1]);
        gru_layer_kernel<GRU_H><<<64, 256, 0, stream>>>(
            hbuf[0][p ^ 1], wih_b[1], whh_b[1], bih[1], bhh[1],
            hbuf[1][p], hbuf[1][p ^ 1]);
        gru_layer_kernel<GRU_H><<<64, 256, 0, stream>>>(
            hbuf[1][p ^ 1], wih_b[2], whh_b[2], bih[2], bhh[2],
            hbuf[2][p], hbuf[2][p ^ 1]);
        fc_argmax_kernel<<<GRU_B, GRU_V, 0, stream>>>(
            hbuf[2][p ^ 1], fcw, fcb, out, x0, t);
    }
}
